// HeteroGNN_22557168239386
// MI455X (gfx1250) — matrix operand order repaired and checked by reading, not compiled, not run on hardware
//
#include <hip/hip_runtime.h>
#include <math.h>

// ---------------------------------------------------------------------------
// HeteroGNN (2-layer GATv2) forward for gfx1250.
//   N=100000 nodes/type, E=600000 edges/rel, H=128, OUT=64.
//   Output depends only on layer0 {rev,writes,likes} + layer1 rev + final linear;
//   l1_w / l1_lk are dead code and skipped.
// GEMMs: LDS-staged f16 tiles -> v_wmma_f32_16x16x32_f16 (f32 accumulate),
// fully unrolled per (K, NOUT) template instantiation.
// ---------------------------------------------------------------------------

#define NN 100000
#define EE 600000
#define HH 128
#define OUTC 64
#define NEG_SLOPE 0.2f

typedef __attribute__((ext_vector_type(16))) _Float16 v16h;
typedef __attribute__((ext_vector_type(8)))  _Float16 v8h;
typedef __attribute__((ext_vector_type(8)))  float    v8f;

// order-preserving uint encoding for float atomic max
__device__ __forceinline__ unsigned enc_f(float f) {
  unsigned u = __float_as_uint(f);
  return (u & 0x80000000u) ? ~u : (u | 0x80000000u);
}
__device__ __forceinline__ float dec_f(unsigned u) {
  return (u & 0x80000000u) ? __uint_as_float(u & 0x7fffffffu)
                           : __uint_as_float(~u);
}

// ---------------------------------------------------------------------------
// C[M x NOUT] = A[M x K] @ B[K x NOUT] + bias.
// One block (256 thr = 8 waves) per 16-row M tile; wave w owns columns
// [16w, 16w+16).  A-chunk (16 x CK) and transposed B-chunk (NOUT x CK) staged
// in LDS as f16 with +8-half padding (all fragment reads 16B-aligned,
// bank-conflict-free strides).  K-loop fully unrolled -> dense WMMA chain.
// ---------------------------------------------------------------------------
template <int K, int NOUT>
__global__ void __launch_bounds__(256)
wmma_gemm_lds(const float* __restrict__ A, const float* __restrict__ B,
              const float* __restrict__ bias, float* __restrict__ C) {
  constexpr int CK  = (K < 128) ? K : 128;  // K-chunk resident in LDS
  constexpr int LDA = CK + 8;               // padded strides (halves)
  constexpr int LDB = CK + 8;
  __shared__ _Float16 Ah[16 * LDA];
  __shared__ _Float16 Bt[NOUT * LDB];

  const int tid  = threadIdx.x;
  const int lane = tid & 31;
  const int wv   = tid >> 5;                 // 0..7
  const int tm   = blockIdx.x;               // 16-row tile
  const int row  = lane & 15;                // A row within tile (both halves)
  const int col  = wv * 16 + (lane & 15);    // this wave's output column
  const bool active = (wv * 16) < NOUT;

  v8f acc = {};
#pragma unroll
  for (int k0 = 0; k0 < K; k0 += CK) {
    // ---- stage A chunk: Ah[r][k] = A[tm*16+r][k0+k] ----
#pragma unroll 4
    for (int i = tid; i < 16 * CK; i += 256) {
      const int r = i / CK, k = i % CK;
      Ah[r * LDA + k] = (_Float16)A[((size_t)tm * 16 + r) * K + k0 + k];
    }
    // ---- stage B chunk transposed: Bt[n][k] = B[k0+k][n] ----
#pragma unroll 4
    for (int i = tid; i < CK * NOUT; i += 256) {
      const int k = i / NOUT, n = i % NOUT;
      Bt[n * LDB + k] = (_Float16)B[(size_t)(k0 + k) * NOUT + n];
    }
    __syncthreads();

    if (active) {
#pragma unroll
      for (int kk = 0; kk < CK; kk += 32) {
        // A frag: lanes 0-15 -> K {kk..kk+7, kk+16..kk+23};
        //         lanes 16-31 -> K {kk+8..kk+15, kk+24..kk+31}
        const int ka = kk + ((lane < 16) ? 0 : 8);
        const _Float16* ap = &Ah[row * LDA + ka];
        const v8h a0 = *(const v8h*)ap;
        const v8h a1 = *(const v8h*)(ap + 16);
        v16h a;
#pragma unroll
        for (int j = 0; j < 8; ++j) { a[j] = a0[j]; a[8 + j] = a1[j]; }

        // B frag: lanes 0-15 -> K {kk..kk+7, kk+16..kk+23} (col=lane); lanes 16-31 -> +8
        const int kb = kk + ((lane < 16) ? 0 : 8);
        const _Float16* bp = &Bt[col * LDB + kb];
        const v8h b0 = *(const v8h*)bp;
        const v8h b1 = *(const v8h*)(bp + 16);
        v16h b;
#pragma unroll
        for (int j = 0; j < 8; ++j) { b[j] = b0[j]; b[8 + j] = b1[j]; }

        acc = __builtin_amdgcn_wmma_f32_16x16x32_f16(
            false, a, false, b, (short)0, acc, false, false);
      }
    }
    __syncthreads();
  }

  if (active) {
    const float bv = bias[col];
    // C layout: lanes 0-15 hold rows 0..7, lanes 16-31 rows 8..15
    const int rbase = tm * 16 + ((lane < 16) ? 0 : 8);
#pragma unroll
    for (int r = 0; r < 8; ++r)
      C[(size_t)(rbase + r) * NOUT + col] = acc[r] + bv;
  }
}

static void launch_gemm(const float* A, const float* B, const float* bias,
                        float* C, int K, int Nout, hipStream_t st) {
  const dim3 g(NN / 16);
  if (Nout == OUTC) { wmma_gemm_lds<HH, OUTC><<<g, 256, 0, st>>>(A, B, bias, C); return; }
  switch (K) {
    case 32:  wmma_gemm_lds<32,  HH><<<g, 256, 0, st>>>(A, B, bias, C); break;
    case 64:  wmma_gemm_lds<64,  HH><<<g, 256, 0, st>>>(A, B, bias, C); break;
    case 128: wmma_gemm_lds<128, HH><<<g, 256, 0, st>>>(A, B, bias, C); break;
    default:  wmma_gemm_lds<256, HH><<<g, 256, 0, st>>>(A, B, bias, C); break;
  }
}

// ---------------------------------------------------------------------------
__global__ void init_seg(unsigned* __restrict__ emax, float* __restrict__ denom, int n) {
  int i = blockIdx.x * blockDim.x + threadIdx.x;
  if (i < n) { emax[i] = 0u; denom[i] = 0.0f; }
}

// dest[n*ld + coff + h] = bias[h]
__global__ void bias_init(float* __restrict__ dest, int ld, int coff,
                          const float* __restrict__ bias, int n) {
  int i = blockIdx.x * blockDim.x + threadIdx.x;
  if (i >= n * HH) return;
  int node = i >> 7, h = i & 127;
  dest[(size_t)node * ld + coff + h] = bias[h];
}

// pass 1: one wave per edge; score + segment-max (order-preserving uint atomicMax)
__global__ void __launch_bounds__(256)
edge_score(const float* __restrict__ xl, const float* __restrict__ xr,
           const float* __restrict__ ea, int de,
           const float* __restrict__ We, const float* __restrict__ att,
           const int* __restrict__ ei, float* __restrict__ score,
           unsigned* __restrict__ emax) {
  const int e = (blockIdx.x * blockDim.x + threadIdx.x) >> 5;
  const int lane = threadIdx.x & 31;
  if (e >= EE) return;
  const int src = ei[e];
  const int dst = ei[EE + e];

  float eav[5];
  for (int k = 0; k < de; ++k) eav[k] = ea[(size_t)e * de + k];

  float acc = 0.0f;
#pragma unroll
  for (int hh = 0; hh < 4; ++hh) {
    const int h = lane + hh * 32;
    float m = xl[(size_t)src * HH + h] + xr[(size_t)dst * HH + h];
    for (int k = 0; k < de; ++k) m += eav[k] * We[k * HH + h];
    const float lr = (m > 0.0f) ? m : NEG_SLOPE * m;
    acc += lr * att[h];
  }
#pragma unroll
  for (int off = 16; off > 0; off >>= 1) acc += __shfl_xor(acc, off, 32);

  if (lane == 0) {
    score[e] = acc;
    atomicMax(&emax[dst], enc_f(acc));
  }
}

// pass 2: w = exp(score - segmax[dst]); denom[dst] += w
__global__ void edge_norm(const float* __restrict__ score, float* __restrict__ wexp,
                          const int* __restrict__ ei, const unsigned* __restrict__ emax,
                          float* __restrict__ denom) {
  int e = blockIdx.x * blockDim.x + threadIdx.x;
  if (e >= EE) return;
  const int dst = ei[EE + e];
  const float w = __expf(score[e] - dec_f(emax[dst]));
  wexp[e] = w;
  atomicAdd(&denom[dst], w);
}

// pass 3: one block (128 thr) per edge; dest[dst][coff+h] += alpha * xl[src][h]
__global__ void __launch_bounds__(128)
edge_aggregate(const float* __restrict__ xl, const int* __restrict__ ei,
               const float* __restrict__ wexp, const float* __restrict__ denom,
               float* __restrict__ dest, int ld, int coff) {
  const int e = blockIdx.x;
  const int h = threadIdx.x;
  const int src = ei[e];
  const int dst = ei[EE + e];
  const float d = denom[dst];
  const float alpha = wexp[e] / ((d > 0.0f) ? d : 1.0f);
  atomicAdd(&dest[(size_t)dst * ld + coff + h], alpha * xl[(size_t)src * HH + h]);
}

__global__ void relu_inplace(float* __restrict__ x, long long count) {
  long long i = (long long)blockIdx.x * blockDim.x + threadIdx.x;
  if (i < count) x[i] = fmaxf(x[i], 0.0f);
}

// ---------------------------------------------------------------------------
struct Scratch {
  float *xl, *xr, *bufA1, *bufP1, *bufA2, *score, *wexp, *denom;
  unsigned* emax;
};

static void run_conv(const float* xsrc, int dsrc, const float* xdst, int ddst,
                     const int* ei, const float* ea, int de,
                     const float* Wl, const float* bl,
                     const float* Wr, const float* br,
                     const float* We, const float* att, const float* b,
                     float* dest, int dld, int dcoff,
                     const Scratch& s, hipStream_t st) {
  launch_gemm(xsrc, Wl, bl, s.xl, dsrc, HH, st);   // xl = xsrc @ Wl + bl
  launch_gemm(xdst, Wr, br, s.xr, ddst, HH, st);   // xr = xdst @ Wr + br
  init_seg<<<(NN + 255) / 256, 256, 0, st>>>(s.emax, s.denom, NN);
  bias_init<<<((size_t)NN * HH + 255) / 256, 256, 0, st>>>(dest, dld, dcoff, b, NN);
  edge_score<<<(EE + 7) / 8, 256, 0, st>>>(s.xl, s.xr, ea, de, We, att, ei, s.score, s.emax);
  edge_norm<<<(EE + 255) / 256, 256, 0, st>>>(s.score, s.wexp, ei, s.emax, s.denom);
  edge_aggregate<<<EE, 128, 0, st>>>(s.xl, ei, s.wexp, s.denom, dest, dld, dcoff);
}

extern "C" void kernel_launch(void* const* d_in, const int* in_sizes, int n_in,
                              void* d_out, int out_size, void* d_ws, size_t ws_size,
                              hipStream_t stream) {
  (void)in_sizes; (void)n_in; (void)out_size; (void)ws_size;

  // ---- input map (setup_inputs() insertion order; params flattened in dict
  //      insertion order: {l0_w,l0_lk,l0_rw,l1_w,l1_lk,l1_rw} x {Wl,bl,Wr,br,We,att,b},
  //      then lin_W, lin_b) ----
  const float* x_author = (const float*)d_in[0];   // [N,64]
  const float* x_paper  = (const float*)d_in[1];   // [N,32]
  const float* ea_w  = (const float*)d_in[2];      // [E,3]
  const float* ea_lk = (const float*)d_in[3];      // [E,5]
  const float* ea_rv = (const float*)d_in[4];      // [E,4]
  auto P = [&](int rel, int leaf) { return (const float*)d_in[5 + rel * 7 + leaf]; };
  // leaf: 0=Wl 1=bl 2=Wr 3=br 4=We 5=att 6=b ; rel: 0=l0_w 1=l0_lk 2=l0_rw 3=l1_w 4=l1_lk 5=l1_rw
  const float* lin_W = (const float*)d_in[47];     // [128,64]
  const float* lin_b = (const float*)d_in[48];     // [64]
  const int* ei_w  = (const int*)d_in[49];         // [2,E]
  const int* ei_lk = (const int*)d_in[50];
  const int* ei_rv = (const int*)d_in[51];

  // ---- workspace carve-up (floats) ----
  float* base = (float*)d_ws;
  Scratch s;
  s.xl    = base;                         base += (size_t)NN * HH;
  s.xr    = base;                         base += (size_t)NN * HH;
  s.bufA1 = base;                         base += (size_t)NN * HH;      // author after L0
  s.bufP1 = base;                         base += (size_t)NN * 2 * HH;  // paper  after L0 (cat)
  s.bufA2 = base;                         base += (size_t)NN * HH;      // author after L1
  s.score = base;                         base += EE;
  s.wexp  = base;                         base += EE;
  s.denom = base;                         base += NN;
  s.emax  = (unsigned*)base;

  // ================= layer 0 =================
  // author <- rev(paper)      (l0_rw: Wl paper(32), Wr author(64), We 4)
  run_conv(x_paper, 32, x_author, 64, ei_rv, ea_rv, 4,
           P(2,0), P(2,1), P(2,2), P(2,3), P(2,4), P(2,5), P(2,6),
           s.bufA1, HH, 0, s, stream);
  // paper[:,0:128] <- writes(author)   (l0_w: Wl author(64), Wr paper(32), We 3)
  run_conv(x_author, 64, x_paper, 32, ei_w, ea_w, 3,
           P(0,0), P(0,1), P(0,2), P(0,3), P(0,4), P(0,5), P(0,6),
           s.bufP1, 2 * HH, 0, s, stream);
  // paper[:,128:256] <- likes(author)  (l0_lk: We 5)
  run_conv(x_author, 64, x_paper, 32, ei_lk, ea_lk, 5,
           P(1,0), P(1,1), P(1,2), P(1,3), P(1,4), P(1,5), P(1,6),
           s.bufP1, 2 * HH, HH, s, stream);
  relu_inplace<<<((long long)NN * HH + 255) / 256, 256, 0, stream>>>(s.bufA1, (long long)NN * HH);
  relu_inplace<<<((long long)NN * 2 * HH + 255) / 256, 256, 0, stream>>>(s.bufP1, (long long)NN * 2 * HH);

  // ================= layer 1 (only the rev conv feeds the output) =================
  // author <- rev(paper)  (l1_rw: Wl paper(256), Wr author(128), We 4)
  run_conv(s.bufP1, 2 * HH, s.bufA1, HH, ei_rv, ea_rv, 4,
           P(5,0), P(5,1), P(5,2), P(5,3), P(5,4), P(5,5), P(5,6),
           s.bufA2, HH, 0, s, stream);
  relu_inplace<<<((long long)NN * HH + 255) / 256, 256, 0, stream>>>(s.bufA2, (long long)NN * HH);

  // ================= final linear: out = A2 @ lin_W + lin_b  [N,64] =================
  launch_gemm(s.bufA2, lin_W, lin_b, (float*)d_out, HH, OUTC, stream);
}
